// CausalSelfAttention_8521215115405
// MI455X (gfx1250) — matrix operand order repaired and checked by reading, not compiled, not run on hardware
//
#include <hip/hip_runtime.h>
#include <cstdint>
#include <cstddef>

// ---------------------------------------------------------------------------
// Causal self-attention, CDNA5 (gfx1250), wave32, bf16 WMMA with f32 accum.
// S=4096, D=1024, H=16, HD=64. Compute-bound (~103 GFLOP; all operands are
// L2-resident vs 192MB / 23.3TB/s), so all matmuls use v_wmma_f32_16x16x32_bf16.
// Async global->LDS (ASYNCcnt) is used where operands are already bf16,
// guarded by __has_builtin so compilation never regresses.
// ---------------------------------------------------------------------------

#define S_LEN 4096
#define D_MODEL 1024
#define N_HEADS 16
#define HEAD_DIM 64

typedef __attribute__((ext_vector_type(16))) __bf16 v16bf;
typedef __attribute__((ext_vector_type(8)))  __bf16 v8bf;
typedef __attribute__((ext_vector_type(4)))  __bf16 v4bf;
typedef __attribute__((ext_vector_type(8)))  float  v8f;

// Builtin prototype (from hipcc diagnostic): params are pointers to
// GCC-style vector_size(16) int, with global (AS1) / LDS (AS3) qualifiers.
typedef int v4i_raw __attribute__((vector_size(16)));
typedef __attribute__((address_space(1))) v4i_raw as1_v4i;
typedef __attribute__((address_space(3))) v4i_raw as3_v4i;

#if defined(__AMDGCN__) && __has_builtin(__builtin_amdgcn_global_load_async_to_lds_b128)
#define HAVE_ASYNC_LDS 1
#endif

__device__ inline __bf16 f2bf(float f) {
  unsigned u = __builtin_bit_cast(unsigned, f);
  unsigned short h = (unsigned short)((u + 0x7FFFu + ((u >> 16) & 1u)) >> 16);
  return __builtin_bit_cast(__bf16, h);
}

__device__ inline v8bf ld8(const __bf16* p) {   // 16B load (ds_load_b128 / global_load_b128)
  return *reinterpret_cast<const v8bf*>(p);
}

__device__ inline v16bf mk16(v8bf lo, v8bf hi) {
  union { v16bf v; v8bf h[2]; } u;
  u.h[0] = lo; u.h[1] = hi;
  return u.v;
}

__device__ inline v8f wmma_bf16(v16bf a, v16bf b, v8f c) {
  return __builtin_amdgcn_wmma_f32_16x16x32_bf16(false, a, false, b, (short)0, c, false, false);
}

// 16-byte global -> LDS copy: async (ASYNCcnt) when available, else via VGPRs.
__device__ inline void copy16_g2l(const __bf16* g, __bf16* l) {
#if defined(HAVE_ASYNC_LDS)
  __builtin_amdgcn_global_load_async_to_lds_b128(
      (as1_v4i*)(void*)(uintptr_t)g, (as3_v4i*)(void*)l, 0, 0);
#else
  *(v8bf*)l = ld8(g);
#endif
}

template<int N> __device__ inline void wait_async() {
#if defined(HAVE_ASYNC_LDS)
#if __has_builtin(__builtin_amdgcn_s_wait_asynccnt)
  __builtin_amdgcn_s_wait_asynccnt(N);
#else
  asm volatile("s_wait_asynccnt %0" :: "i"(N) : "memory");
#endif
#endif
}

// ---------------------------------------------------------------------------
// NT GEMM: C[m,n] = sum_k A[m,k] * B[n,k]   (exactly einsum('bsd,ed->bse'))
// BM=BN=128, BK=64; 256 threads = 8 waves; wave -> 32x64 (2x4 WMMA tiles),
// 16 WMMAs per wave per stage (one barrier pair per 64 K-steps).
// AMODE: 0 = A fp32 (convert to bf16), 1 = A bf16 (async-copy eligible).
// EPI:   0 = bf16 store, 1 = RoPE + bf16 store, 2 = fp32 store.
// ---------------------------------------------------------------------------
template<int AMODE, int EPI>
__global__ __launch_bounds__(256) void gemm_nt_kernel(const void* __restrict__ Ain,
                                                      const float* __restrict__ Bw,
                                                      void* __restrict__ Cout,
                                                      int M, int N, int K) {
  constexpr int BM = 128, BK = 64, ASTR = 72;  // 72-half stride: 144B rows, 16B-aligned frags
  __shared__ __align__(16) __bf16 As[BM * ASTR];
  __shared__ __align__(16) __bf16 Bs[BM * ASTR];

  const int row0 = blockIdx.y * BM;
  const int col0 = blockIdx.x * BM;
  const int t    = threadIdx.x;
  const int wave = t >> 5, lane = t & 31;
  const int wr = wave & 3, wc = wave >> 2;     // 4x2 wave grid -> 128x128
  const int lcol = lane & 15, g = lane >> 4;

  v8f acc[2][4] = {};

  for (int k0 = 0; k0 < K; k0 += BK) {
    // ---- B tile: fp32 -> bf16 staging (16 cols x 4 floats, 16 rows x 8) ----
    {
      const int tc = t & 15, tr = t >> 4;
      for (int rr = 0; rr < 8; ++rr) {
        const int r = tr + rr * 16;
        const float4 fb = *(const float4*)(Bw + (size_t)(col0 + r) * K + k0 + tc * 4);
        v4bf pkb = { f2bf(fb.x), f2bf(fb.y), f2bf(fb.z), f2bf(fb.w) };
        *(v4bf*)(&Bs[r * ASTR + tc * 4]) = pkb;
      }
    }
    // ---- A tile ----
    if (AMODE == 0) {
      const int tc = t & 15, tr = t >> 4;
      for (int rr = 0; rr < 8; ++rr) {
        const int r = tr + rr * 16;
        const float4 f = *(const float4*)((const float*)Ain + (size_t)(row0 + r) * K + k0 + tc * 4);
        v4bf pk = { f2bf(f.x), f2bf(f.y), f2bf(f.z), f2bf(f.w) };
        *(v4bf*)(&As[r * ASTR + tc * 4]) = pk;
      }
    } else {
      const int tc = t & 7, tr = t >> 3;       // 8 x 16B chunks per row, 32 rows
      for (int rr = 0; rr < 4; ++rr) {
        const int r = tr + rr * 32;
        const __bf16* src = (const __bf16*)Ain + (size_t)(row0 + r) * K + k0 + tc * 8;
        copy16_g2l(src, &As[r * ASTR + tc * 8]);
      }
    }
    if (AMODE == 1) wait_async<0>();
    __syncthreads();

    // ---- fragments per documented wave32 layouts; 2 K-steps of 32 ----
    for (int kk = 0; kk < 2; ++kk) {
      v16bf af[2], bfr[4];
      for (int mt = 0; mt < 2; ++mt) {
        const __bf16* ab = &As[(wr * 32 + mt * 16 + lcol) * ASTR + kk * 32];
        af[mt] = mk16(ld8(ab + g * 8), ld8(ab + 16 + g * 8));
      }
      for (int nt = 0; nt < 4; ++nt) {
        const __bf16* bb = &Bs[(wc * 64 + nt * 16 + lcol) * ASTR + kk * 32];
        bfr[nt] = mk16(ld8(bb + g * 8), ld8(bb + 16 + g * 8));
      }
      for (int mt = 0; mt < 2; ++mt)
        for (int nt = 0; nt < 4; ++nt)
          acc[mt][nt] = wmma_bf16(af[mt], bfr[nt], acc[mt][nt]);
    }
    __syncthreads();
  }

  // ---- epilogue: C layout VGPR r -> M=r+8*g, N=lcol ----
  for (int mt = 0; mt < 2; ++mt)
    for (int nt = 0; nt < 4; ++nt)
      for (int r = 0; r < 8; ++r) {
        const int row = row0 + wr * 32 + mt * 16 + r + 8 * g;
        const int col = col0 + wc * 64 + nt * 16 + lcol;
        float v = acc[mt][nt][r];
        if (EPI == 1) {
          // RoPE: pair (even,odd) along head dim = adjacent lanes in N.
          const int dh = col & (HEAD_DIM - 1);
          const float freq = __expf(-(float)(dh & 0x3E) * (9.2103403720f / 64.0f)); // theta^(-2p/64)
          const float ang  = (float)row * freq;
          const float cs = __cosf(ang), sn = __sinf(ang);
          const float partner = __shfl_xor(v, 1, 32);
          v = (dh & 1) ? (partner * sn + v * cs) : (v * cs - partner * sn);
        }
        if (EPI == 2) {
          ((float*)Cout)[(size_t)row * N + col] = v;
        } else {
          ((__bf16*)Cout)[(size_t)row * N + col] = f2bf(v);
        }
      }
}

// ---------------------------------------------------------------------------
// Flash-style causal attention. Grid: (S/128, H). WG = 8 waves x 16 q-rows.
// Key blocks of 64: K staged with double-buffered async global->LDS copies,
// V staged transposed via VGPRs (so P*V B-fragments read contiguously).
// Online softmax in f32; P re-laid out C->A fragment via per-wave LDS.
// ---------------------------------------------------------------------------
__global__ __launch_bounds__(256) void attn_kernel(const void* __restrict__ Qv,
                                                   const void* __restrict__ Kv,
                                                   const void* __restrict__ Vv,
                                                   void* __restrict__ Ov) {
  constexpr int D = D_MODEL, HD = HEAD_DIM, KSTR = 72, PSTR = 72;
  __shared__ __align__(16) __bf16 Ks[2][64 * KSTR];
  __shared__ __align__(16) __bf16 Vt[2][64 * KSTR];   // transposed: Vt[d][k]
  __shared__ __align__(16) __bf16 Ps[8 * 16 * PSTR];

  const __bf16* Qb = (const __bf16*)Qv;
  const __bf16* Kb = (const __bf16*)Kv;
  const __bf16* Vb = (const __bf16*)Vv;
  __bf16*       Ob = (__bf16*)Ov;

  const int h    = blockIdx.y;
  const int t    = threadIdx.x;
  const int wave = t >> 5, lane = t & 31;
  const int lcol = lane & 15, g = lane >> 4;
  const int q0   = blockIdx.x * 128 + wave * 16;

  // Per-stage: each wave issues 2 async K copies (ASYNCcnt += 2).
  auto stage = [&](int kb, int buf) {
    const int tc = t & 7, tr = t >> 3;
    for (int rr = 0; rr < 2; ++rr) {
      const int r = tr + rr * 32;
      const __bf16* kp = Kb + (size_t)(kb * 64 + r) * D + h * HD + tc * 8;
      copy16_g2l(kp, &Ks[buf][r * KSTR + tc * 8]);
      const __bf16* vp = Vb + (size_t)(kb * 64 + r) * D + h * HD + tc * 8;
      v8bf vv = ld8(vp);
      for (int j = 0; j < 8; ++j) Vt[buf][(tc * 8 + j) * KSTR + r] = vv[j];
    }
  };

  // Q fragments (A layout), loaded once: 16 rows x 64 head-dims
  v16bf aq[2];
  {
    const __bf16* qp = Qb + (size_t)(q0 + lcol) * D + h * HD;
    for (int kk = 0; kk < 2; ++kk)
      aq[kk] = mk16(ld8(qp + kk * 32 + g * 8), ld8(qp + kk * 32 + 16 + g * 8));
  }

  float mrow[8], lrow[8];
  v8f o[4] = {};
  for (int r = 0; r < 8; ++r) { mrow[r] = -1e30f; lrow[r] = 0.0f; }

  const int nkb = 2 * blockIdx.x + 2;            // causal loop bound
  stage(0, 0);
  for (int kb = 0; kb < nkb; ++kb) {
    const int cur  = kb & 1;
    const bool more = (kb + 1) < nkb;
    if (more) stage(kb + 1, cur ^ 1);            // prefetch next block
    if (more) wait_async<2>(); else wait_async<0>();  // in-order: block kb is done
    __syncthreads();

    if (kb * 64 <= q0 + 15) {                    // wave-uniform: EXEC stays all-1s
      // S = Q K^T (scaled), 4 N-tiles x K=64
      v8f s[4];
      for (int nt = 0; nt < 4; ++nt) {
        v8f c = {};
        for (int kk = 0; kk < 2; ++kk) {
          const __bf16* bb = &Ks[cur][(nt * 16 + lcol) * KSTR + kk * 32];
          c = wmma_bf16(aq[kk], mk16(ld8(bb + g * 8), ld8(bb + 16 + g * 8)), c);
        }
        s[nt] = c;
      }
      // scale + causal mask
      for (int nt = 0; nt < 4; ++nt)
        for (int r = 0; r < 8; ++r) {
          const int key = kb * 64 + nt * 16 + lcol;
          const int qr  = q0 + r + 8 * g;
          const float sv = s[nt][r] * 0.125f;    // 1/sqrt(64)
          s[nt][r] = (key > qr) ? -1e30f : sv;
        }
      // online softmax (row = C-fragment VGPR index; reduce across 16-lane group)
      for (int r = 0; r < 8; ++r) {
        float mx = s[0][r];
        for (int nt = 1; nt < 4; ++nt) mx = fmaxf(mx, s[nt][r]);
        for (int off = 1; off < 16; off <<= 1) mx = fmaxf(mx, __shfl_xor(mx, off, 32));
        const float mnew  = fmaxf(mrow[r], mx);
        const float alpha = __expf(mrow[r] - mnew);
        mrow[r] = mnew;
        float rs = 0.0f;
        for (int nt = 0; nt < 4; ++nt) {
          const float p = __expf(s[nt][r] - mnew);
          s[nt][r] = p;
          rs += p;
        }
        for (int off = 1; off < 16; off <<= 1) rs += __shfl_xor(rs, off, 32);
        lrow[r] = lrow[r] * alpha + rs;
        for (int nt = 0; nt < 4; ++nt) o[nt][r] *= alpha;
      }
      // P: C layout -> per-wave LDS -> A fragments (bf16)
      __bf16* pw = &Ps[wave * 16 * PSTR];
      for (int nt = 0; nt < 4; ++nt)
        for (int r = 0; r < 8; ++r)
          pw[(r + 8 * g) * PSTR + nt * 16 + lcol] = f2bf(s[nt][r]);
      v16bf ap[2];
      const __bf16* pb = pw + lcol * PSTR;
      for (int kk = 0; kk < 2; ++kk)
        ap[kk] = mk16(ld8(pb + kk * 32 + g * 8), ld8(pb + kk * 32 + 16 + g * 8));
      // O += P V  (B frags from transposed V: contiguous along k)
      for (int nt = 0; nt < 4; ++nt)
        for (int kk = 0; kk < 2; ++kk) {
          const __bf16* bb = &Vt[cur][(nt * 16 + lcol) * KSTR + kk * 32];
          o[nt] = wmma_bf16(ap[kk], mk16(ld8(bb + g * 8), ld8(bb + 16 + g * 8)), o[nt]);
        }
    }
    __syncthreads();
  }

  // normalize + store bf16 attention output [s, h*64+d]
  for (int nt = 0; nt < 4; ++nt)
    for (int r = 0; r < 8; ++r) {
      const int qr = q0 + r + 8 * g;
      const float val = o[nt][r] / lrow[r];
      Ob[(size_t)qr * D + h * HD + nt * 16 + lcol] = f2bf(val);
    }
}

// ---------------------------------------------------------------------------
extern "C" void kernel_launch(void* const* d_in, const int* in_sizes, int n_in,
                              void* d_out, int out_size, void* d_ws, size_t ws_size,
                              hipStream_t stream) {
  const float* x  = (const float*)d_in[0];
  const float* Wq = (const float*)d_in[1];
  const float* Wk = (const float*)d_in[2];
  const float* Wv = (const float*)d_in[3];
  const float* Wo = (const float*)d_in[4];

  const size_t seg = (size_t)S_LEN * D_MODEL * sizeof(unsigned short); // 8 MB each
  char* ws = (char*)d_ws;
  void* Qb = (void*)(ws + 0 * seg);
  void* Kb = (void*)(ws + 1 * seg);
  void* Vb = (void*)(ws + 2 * seg);
  void* Ab = (void*)(ws + 3 * seg);

  const dim3 gg(D_MODEL / 128, S_LEN / 128);   // (8, 32)
  const dim3 bb(256);

  // Q/K projections with fused RoPE; V projection plain. All stored bf16.
  gemm_nt_kernel<0, 1><<<gg, bb, 0, stream>>>(x, Wq, Qb, S_LEN, D_MODEL, D_MODEL);
  gemm_nt_kernel<0, 1><<<gg, bb, 0, stream>>>(x, Wk, Kb, S_LEN, D_MODEL, D_MODEL);
  gemm_nt_kernel<0, 0><<<gg, bb, 0, stream>>>(x, Wv, Vb, S_LEN, D_MODEL, D_MODEL);

  // Causal flash attention per (q-block, head).
  attn_kernel<<<dim3(S_LEN / 128, N_HEADS), bb, 0, stream>>>(Qb, Kb, Vb, Ab);

  // Output projection -> fp32 d_out (A operand async-copied to LDS).
  gemm_nt_kernel<1, 2><<<gg, bb, 0, stream>>>(Ab, Wo, d_out, S_LEN, D_MODEL, D_MODEL);
}
